// CSBrainBackbone_67319317398062
// MI455X (gfx1250) — matrix operand order repaired and checked by reading, not compiled, not run on hardware
//
#include <hip/hip_runtime.h>
#include <math.h>

// ---------------- model dimensions ----------------
#define Bsz   16
#define Cn    19
#define Tn    64
#define Dm    200
#define ROWSn 1216            // Cn*Tn
#define TOKn  19456           // Bsz*Cn*Tn
#define TOKD  3891200L        // TOKn*200
#define WSn   5
#define NWn   13
#define SWINn 1520            // Bsz*WSn*Cn
#define TOKWn 19760           // SWINn*NWn
#define SREGn 1024            // Bsz*Tn
#define DFFn  800
#define SPLD  104             // padded row stride for the [TOK,101] spectrum (8B-aligned rows)

typedef __attribute__((ext_vector_type(16))) _Float16 v16h;
typedef __attribute__((ext_vector_type(8)))  float    v8f;

__constant__ int kRegStart[6] = {0, 3, 7, 12, 16, 19};

struct BrainParams {
  const float* w[5][3];
  const float* b[5][3];
};

__device__ __forceinline__ float gelu_exact(float x) {
  return 0.5f * x * (1.0f + erff(x * 0.70710678118654752440f));
}

// ==================== WMMA GEMM ====================
// C[M,N] = epi( A[M,K](lda) @ W  (+bias) (+resid) (GELU?) )
// BT=true : W stored [N,K] row-major   (torch Linear: x @ w.T)
// BT=false: W stored [K,N] row-major   (x @ w)
// One wave -> 32x32 f32 tile = 4 x v_wmma_f32_16x16x32_f16 per 32-wide K step.
// Main K loop is unguarded (vector float2 loads); tail is branchless
// (clamped addresses + value select) so EXEC stays all-ones for WMMA.
template <bool BT>
__global__ void gemm_wmma_f16(const float* __restrict__ A, int lda,
                              const float* __restrict__ W,
                              const float* __restrict__ bias, const float* __restrict__ resid,
                              float* __restrict__ C, int M, int N, int K, int doGelu) {
  const int lane  = threadIdx.x & 31;
  const int mtile = (blockIdx.y * blockDim.y + threadIdx.y) * 32;
  const int ntile = blockIdx.x * 32;
  if (mtile >= M) return;                 // wave-uniform
  const int half = lane >> 4;
  const int l16  = lane & 15;
  int am0 = mtile + l16;       if (am0 >= M) am0 = M - 1;   // clamp: safe duplicate reads
  int am1 = mtile + 16 + l16;  if (am1 >= M) am1 = M - 1;
  const int n0  = ntile + l16;
  const int n1  = ntile + 16 + l16;
  const int bn0 = (n0 < N) ? n0 : N - 1;
  const int bn1 = (n1 < N) ? n1 : N - 1;
  const float* __restrict__ Ar0 = A + (size_t)am0 * lda;
  const float* __restrict__ Ar1 = A + (size_t)am1 * lda;
  const float* __restrict__ Wn0 = BT ? (W + (size_t)bn0 * K) : (W + bn0);
  const float* __restrict__ Wn1 = BT ? (W + (size_t)bn1 * K) : (W + bn1);
  v8f acc00 = {}, acc01 = {}, acc10 = {}, acc11 = {};

  const int kFull = K & ~31;
  for (int k0 = 0; k0 < kFull; k0 += 32) {
    v16h a0, a1, b0, b1;
#pragma unroll
    for (int p = 0; p < 8; ++p) {
      // A-matrix 16x32 f16 layout: lanes 0-15 hold K {0..7,16..23}, lanes 16-31 {8..15,24..31}
      const int ka = k0 + ((p < 4) ? (2 * p) : (16 + 2 * (p - 4))) + half * 8;
      const float2 fa0 = *(const float2*)(Ar0 + ka);
      const float2 fa1 = *(const float2*)(Ar1 + ka);
      a0[2 * p] = (_Float16)fa0.x; a0[2 * p + 1] = (_Float16)fa0.y;
      a1[2 * p] = (_Float16)fa1.x; a1[2 * p + 1] = (_Float16)fa1.y;
      // B-matrix 32x16 f16 layout: lanes 0-15 hold K {0..7,16..23}, lanes 16-31 {8..15,24..31}
      const int kb = k0 + ((p < 4) ? (2 * p) : (16 + 2 * (p - 4))) + half * 8;
      if (BT) {
        const float2 fb0 = *(const float2*)(Wn0 + kb);
        const float2 fb1 = *(const float2*)(Wn1 + kb);
        b0[2 * p] = (_Float16)fb0.x; b0[2 * p + 1] = (_Float16)fb0.y;
        b1[2 * p] = (_Float16)fb1.x; b1[2 * p + 1] = (_Float16)fb1.y;
      } else {
        b0[2 * p] = (_Float16)Wn0[(size_t)kb * N]; b0[2 * p + 1] = (_Float16)Wn0[(size_t)(kb + 1) * N];
        b1[2 * p] = (_Float16)Wn1[(size_t)kb * N]; b1[2 * p + 1] = (_Float16)Wn1[(size_t)(kb + 1) * N];
      }
    }
    acc00 = __builtin_amdgcn_wmma_f32_16x16x32_f16(false, a0, false, b0, (short)0, acc00, false, false);
    acc01 = __builtin_amdgcn_wmma_f32_16x16x32_f16(false, a0, false, b1, (short)0, acc01, false, false);
    acc10 = __builtin_amdgcn_wmma_f32_16x16x32_f16(false, a1, false, b0, (short)0, acc10, false, false);
    acc11 = __builtin_amdgcn_wmma_f32_16x16x32_f16(false, a1, false, b1, (short)0, acc11, false, false);
  }

  if (kFull < K) {   // branchless masked tail (loads stay unconditional; values selected)
    v16h a0, a1, b0, b1;
#pragma unroll
    for (int p = 0; p < 8; ++p) {
      const int ka  = kFull + ((p < 4) ? (2 * p) : (16 + 2 * (p - 4))) + half * 8;
      const int kaA = (ka     < K) ? ka     : 0;
      const int kaB = (ka + 1 < K) ? ka + 1 : 0;
      float x;
      x = Ar0[kaA]; if (ka     >= K) x = 0.f; a0[2 * p]     = (_Float16)x;
      x = Ar0[kaB]; if (ka + 1 >= K) x = 0.f; a0[2 * p + 1] = (_Float16)x;
      x = Ar1[kaA]; if (ka     >= K) x = 0.f; a1[2 * p]     = (_Float16)x;
      x = Ar1[kaB]; if (ka + 1 >= K) x = 0.f; a1[2 * p + 1] = (_Float16)x;
      const int kb  = kFull + ((p < 4) ? (2 * p) : (16 + 2 * (p - 4))) + half * 8;
      const int kbA = (kb     < K) ? kb     : 0;
      const int kbB = (kb + 1 < K) ? kb + 1 : 0;
      if (BT) {
        x = Wn0[kbA]; if (kb     >= K) x = 0.f; b0[2 * p]     = (_Float16)x;
        x = Wn0[kbB]; if (kb + 1 >= K) x = 0.f; b0[2 * p + 1] = (_Float16)x;
        x = Wn1[kbA]; if (kb     >= K) x = 0.f; b1[2 * p]     = (_Float16)x;
        x = Wn1[kbB]; if (kb + 1 >= K) x = 0.f; b1[2 * p + 1] = (_Float16)x;
      } else {
        x = Wn0[(size_t)kbA * N]; if (kb     >= K) x = 0.f; b0[2 * p]     = (_Float16)x;
        x = Wn0[(size_t)kbB * N]; if (kb + 1 >= K) x = 0.f; b0[2 * p + 1] = (_Float16)x;
        x = Wn1[(size_t)kbA * N]; if (kb     >= K) x = 0.f; b1[2 * p]     = (_Float16)x;
        x = Wn1[(size_t)kbB * N]; if (kb + 1 >= K) x = 0.f; b1[2 * p + 1] = (_Float16)x;
      }
    }
    acc00 = __builtin_amdgcn_wmma_f32_16x16x32_f16(false, a0, false, b0, (short)0, acc00, false, false);
    acc01 = __builtin_amdgcn_wmma_f32_16x16x32_f16(false, a0, false, b1, (short)0, acc01, false, false);
    acc10 = __builtin_amdgcn_wmma_f32_16x16x32_f16(false, a1, false, b0, (short)0, acc10, false, false);
    acc11 = __builtin_amdgcn_wmma_f32_16x16x32_f16(false, a1, false, b1, (short)0, acc11, false, false);
  }

  const float bv0 = bias ? bias[bn0] : 0.f;   // masked by store guard
  const float bv1 = bias ? bias[bn1] : 0.f;
#pragma unroll
  for (int r = 0; r < 8; ++r) {
    const int m0 = mtile + r + half * 8;      // C layout: lanes0-15 -> M=r, lanes16-31 -> M=8+r
    const int m1 = m0 + 16;
    if (m0 < M && n0 < N) {
      float v = acc00[r] + bv0;
      if (resid) v += resid[(size_t)m0 * N + n0];
      if (doGelu) v = gelu_exact(v);
      C[(size_t)m0 * N + n0] = v;
    }
    if (m0 < M && n1 < N) {
      float v = acc01[r] + bv1;
      if (resid) v += resid[(size_t)m0 * N + n1];
      if (doGelu) v = gelu_exact(v);
      C[(size_t)m0 * N + n1] = v;
    }
    if (m1 < M && n0 < N) {
      float v = acc10[r] + bv0;
      if (resid) v += resid[(size_t)m1 * N + n0];
      if (doGelu) v = gelu_exact(v);
      C[(size_t)m1 * N + n0] = v;
    }
    if (m1 < M && n1 < N) {
      float v = acc11[r] + bv1;
      if (resid) v += resid[(size_t)m1 * N + n1];
      if (doGelu) v = gelu_exact(v);
      C[(size_t)m1 * N + n1] = v;
    }
  }
}

static void launch_gemm(bool bt, const float* A, int lda, const float* W, const float* bias,
                        const float* resid, float* C, int M, int N, int K, int gelu,
                        hipStream_t st) {
  dim3 blk(32, 4);                                   // 4 waves x 32 rows = 128 rows / block
  dim3 grd((N + 31) / 32, (M + 127) / 128);
  if (bt) gemm_wmma_f16<true><<<grd, blk, 0, st>>>(A, lda, W, bias, resid, C, M, N, K, gelu);
  else    gemm_wmma_f16<false><<<grd, blk, 0, st>>>(A, lda, W, bias, resid, C, M, N, K, gelu);
}

// ==================== small MHA (softmax attention) ====================
// qkv rows: [S*L, 600] = [q(200) | k(200) | v(200)], nh=8, dh=25, scale=0.2
template <int L>
__global__ void mha_small(const float* __restrict__ qkv, float* __restrict__ out, int S) {
  long idx = (long)blockIdx.x * blockDim.x + threadIdx.x;
  if (idx >= (long)S * L * 8) return;
  const int  h   = (int)(idx % 8);
  const int  lq  = (int)((idx / 8) % L);
  const long seq = idx / (8 * L);
  const float* __restrict__ base = qkv + (size_t)seq * L * 600;
  float q[25];
#pragma unroll
  for (int i = 0; i < 25; ++i) q[i] = base[(size_t)lq * 600 + h * 25 + i];
  float sc[L];
  float mx = -3.402823466e38f;
#pragma unroll
  for (int lk = 0; lk < L; ++lk) {
    const float* kr = base + (size_t)lk * 600 + 200 + h * 25;
    float s = 0.f;
#pragma unroll
    for (int i = 0; i < 25; ++i) s += q[i] * kr[i];
    s *= 0.2f;
    sc[lk] = s;
    mx = s > mx ? s : mx;
  }
  float den = 0.f;
#pragma unroll
  for (int lk = 0; lk < L; ++lk) { float e = expf(sc[lk] - mx); sc[lk] = e; den += e; }
  const float inv = 1.f / den;
  float o[25];
#pragma unroll
  for (int i = 0; i < 25; ++i) o[i] = 0.f;
#pragma unroll
  for (int lk = 0; lk < L; ++lk) {
    const float wgt = sc[lk] * inv;
    const float* vr = base + (size_t)lk * 600 + 400 + h * 25;
#pragma unroll
    for (int i = 0; i < 25; ++i) o[i] += wgt * vr[i];
  }
  float* orow = out + ((size_t)seq * L + lq) * 200 + h * 25;
#pragma unroll
  for (int i = 0; i < 25; ++i) orow[i] = o[i];
}

// ==================== LayerNorm (one wave per row) ====================
__global__ void layernorm_rows(const float* __restrict__ x, const float* __restrict__ sc,
                               const float* __restrict__ bi, float* __restrict__ y,
                               int rows, int D) {
  int wv   = blockIdx.x * (blockDim.x >> 5) + (threadIdx.x >> 5);
  int lane = threadIdx.x & 31;
  if (wv >= rows) return;
  const float* xr = x + (size_t)wv * D;
  float s = 0.f, s2 = 0.f;
  for (int i = lane; i < D; i += 32) { float v = xr[i]; s += v; s2 += v * v; }
#pragma unroll
  for (int o = 16; o > 0; o >>= 1) { s += __shfl_xor(s, o, 32); s2 += __shfl_xor(s2, o, 32); }
  float m = s / D, var = s2 / D - m * m, inv = rsqrtf(var + 1e-5f);
  float* yr = y + (size_t)wv * D;
  for (int i = lane; i < D; i += 32) yr[i] = (xr[i] - m) * inv * sc[i] + bi[i];
}

// ==================== patch embedding ====================
// conv1: xm[16,1,1216,200], w[25,1,1,49], stride(1,25), pad 24 -> h[16,25,1216,8]
__global__ void patch_conv1(const float* __restrict__ x, const float* __restrict__ w,
                            const float* __restrict__ bias, float* __restrict__ out) {
  long idx = (long)blockIdx.x * blockDim.x + threadIdx.x;
  if (idx >= 3891200L) return;
  int w8 = (int)(idx % 8);
  int r  = (int)((idx / 8) % 1216);
  int oc = (int)((idx / 9728) % 25);
  int b_ = (int)(idx / 243200);
  const float* xr = x + ((size_t)b_ * 1216 + r) * 200;
  const float* wk = w + (size_t)oc * 49;
  float s = bias[oc];
  const int p0 = w8 * 25 - 24;
  for (int kw = 0; kw < 49; ++kw) {
    int p = p0 + kw;
    if (p >= 0 && p < 200) s += wk[kw] * xr[p];
  }
  out[idx] = s;
}

// conv2/conv3: [25,25,1,3], pad (0,1), same spatial shape
__global__ void patch_conv3x(const float* __restrict__ h, const float* __restrict__ w,
                             const float* __restrict__ bias, float* __restrict__ out) {
  long idx = (long)blockIdx.x * blockDim.x + threadIdx.x;
  if (idx >= 3891200L) return;
  int w8 = (int)(idx % 8);
  int r  = (int)((idx / 8) % 1216);
  int oc = (int)((idx / 9728) % 25);
  int b_ = (int)(idx / 243200);
  float s = bias[oc];
  for (int ic = 0; ic < 25; ++ic) {
    const float* hr = h + (((size_t)b_ * 25 + ic) * 1216 + r) * 8;
    const float* wk = w + ((size_t)oc * 25 + ic) * 3;
#pragma unroll
    for (int kw = 0; kw < 3; ++kw) {
      int p = w8 + kw - 1;
      if (p >= 0 && p < 8) s += wk[kw] * hr[p];
    }
  }
  out[idx] = s;
}

// GroupNorm(groups=5) stats over [5ch x 1216 x 8] per (b,g); 80 blocks
__global__ void gn_stats(const float* __restrict__ h, float* __restrict__ stats) {
  const int bg = blockIdx.x;
  const int b_ = bg / 5, g = bg % 5;
  const int n = 5 * 1216 * 8;   // 48640
  float s = 0.f, s2 = 0.f;
  for (int i = threadIdx.x; i < n; i += 256) {
    int cc  = i / 9728;
    int rem = i % 9728;
    float v = h[(((size_t)b_ * 25 + (g * 5 + cc)) * 1216) * 8 + rem];
    s += v; s2 += v * v;
  }
  __shared__ float sh[256], sh2[256];
  sh[threadIdx.x] = s; sh2[threadIdx.x] = s2;
  __syncthreads();
  for (int st = 128; st > 0; st >>= 1) {
    if (threadIdx.x < st) { sh[threadIdx.x] += sh[threadIdx.x + st]; sh2[threadIdx.x] += sh2[threadIdx.x + st]; }
    __syncthreads();
  }
  if (threadIdx.x == 0) {
    float m = sh[0] / n;
    stats[bg * 2]     = m;
    stats[bg * 2 + 1] = sh2[0] / n - m * m;
  }
}

__global__ void gn_apply_gelu(float* __restrict__ h, const float* __restrict__ stats,
                              const float* __restrict__ sc, const float* __restrict__ bi) {
  long idx = (long)blockIdx.x * blockDim.x + threadIdx.x;
  if (idx >= 3891200L) return;
  int ch = (int)((idx / 9728) % 25);
  int b_ = (int)(idx / 243200);
  int g = ch / 5;
  float m = stats[(b_ * 5 + g) * 2], v = stats[(b_ * 5 + g) * 2 + 1];
  float y = (h[idx] - m) * rsqrtf(v + 1e-5f) * sc[ch] + bi[ch];
  h[idx] = gelu_exact(y);
}

// pe[b,c,t,ch*8+w] = h[b,ch,c*64+t,w]
__global__ void patch_permute(const float* __restrict__ h, float* __restrict__ pe) {
  long idx = (long)blockIdx.x * blockDim.x + threadIdx.x;
  if (idx >= TOKD) return;
  int d  = (int)(idx % 200);
  int t  = (int)((idx / 200) % 64);
  int c  = (int)((idx / 12800) % 19);
  int b_ = (int)(idx / 243200);
  int ch = d >> 3, w = d & 7;
  pe[idx] = h[(((size_t)b_ * 25 + ch) * 1216 + (c * 64 + t)) * 8 + w];
}

// rDFT twiddles: e^{-2*pi*i*k*n/200}, reduced via (k*n mod 200) for exact angles
__global__ void twiddle_init(float* __restrict__ tw) {
  int idx = blockIdx.x * blockDim.x + threadIdx.x;
  if (idx >= 101 * 200) return;
  int k = idx / 200, nn = idx % 200;
  int ph = (k * nn) % 200;
  float a = -6.2831853071795864769f * (float)ph / 200.0f;
  tw[2 * idx]     = cosf(a);
  tw[2 * idx + 1] = sinf(a);
}

__global__ void dft_mag(const float* __restrict__ x, const float* __restrict__ tw,
                        float* __restrict__ spec) {
  long idx = (long)blockIdx.x * blockDim.x + threadIdx.x;
  if (idx >= (long)TOKn * 101) return;
  int  kb  = (int)(idx % 101);
  long tok = idx / 101;
  const float* xr = x + tok * 200;
  const float* tr = tw + (size_t)kb * 400;
  float re = 0.f, im = 0.f;
  for (int nn = 0; nn < 200; ++nn) { float v = xr[nn]; re += v * tr[2 * nn]; im += v * tr[2 * nn + 1]; }
  spec[tok * SPLD + kb] = sqrtf(re * re + im * im) * 0.005f;   // norm='forward' -> /200
}

// depthwise positional conv (19x7, pad 9,3, groups=200); out = pe + conv(pe) + bias
__global__ void pos_conv_add(const float* __restrict__ pe, const float* __restrict__ w,
                             const float* __restrict__ bias, float* __restrict__ out) {
  long idx = (long)blockIdx.x * blockDim.x + threadIdx.x;
  if (idx >= TOKD) return;
  int d  = (int)(idx % 200);
  int t  = (int)((idx / 200) % 64);
  int c  = (int)((idx / 12800) % 19);
  int b_ = (int)(idx / 243200);
  const float* wk = w + (size_t)d * 133;
  float s = bias[d];
  for (int kh = 0; kh < 19; ++kh) {
    int cc = c + kh - 9;
    if (cc < 0 || cc >= 19) continue;
    const float* pr = pe + (((size_t)b_ * 19 + cc) * 64) * 200 + d;
    for (int kw = 0; kw < 7; ++kw) {
      int tt = t + kw - 3;
      if (tt < 0 || tt >= 64) continue;
      s += wk[kh * 7 + kw] * pr[(size_t)tt * 200];
    }
  }
  out[idx] = pe[idx] + s;
}

// ==================== temporal / brain embeddings ====================
// y = tem_conv(x) + x ; channels [0,100)=k1, [100,150)=k3, [150,200)=k5
__global__ void tem_embed_add(const float* __restrict__ x, float* __restrict__ y,
                              const float* __restrict__ w1, const float* __restrict__ b1,
                              const float* __restrict__ w3, const float* __restrict__ b3,
                              const float* __restrict__ w5, const float* __restrict__ b5) {
  long idx = (long)blockIdx.x * blockDim.x + threadIdx.x;
  if (idx >= TOKD) return;
  int o  = (int)(idx % 200);
  int t  = (int)((idx / 200) % 64);
  int c  = (int)((idx / 12800) % 19);
  int b_ = (int)(idx / 243200);
  const float* w; const float* bb; int row, k;
  if (o < 100)      { w = w1; bb = b1; row = o;       k = 1; }
  else if (o < 150) { w = w3; bb = b3; row = o - 100; k = 3; }
  else              { w = w5; bb = b5; row = o - 150; k = 5; }
  const int pad = (k - 1) / 2;
  float s = bb[row];
  for (int dk = 0; dk < k; ++dk) {
    int tt = t + dk - pad;
    if (tt < 0 || tt >= 64) continue;
    const float* xr = x + (((size_t)b_ * 19 + c) * 64 + tt) * 200;
    const float* wr = w + (size_t)row * 200 * k + dk;
    float acc = 0.f;
    for (int i = 0; i < 200; ++i) acc += xr[i] * wr[(size_t)i * k];
    s += acc;
  }
  y[idx] = s + x[idx];
}

// y = brain_conv(x) + x ; per-region conv over electrodes with circular pad
__global__ void brain_embed_add(const float* __restrict__ x, float* __restrict__ y,
                                BrainParams bp) {
  long idx = (long)blockIdx.x * blockDim.x + threadIdx.x;
  if (idx >= TOKD) return;
  int o  = (int)(idx % 200);
  int t  = (int)((idx / 200) % 64);
  int c  = (int)((idx / 12800) % 19);
  int b_ = (int)(idx / 243200);
  int r = 0;
#pragma unroll
  for (int q = 0; q < 5; ++q) if (c >= kRegStart[q]) r = q;
  const int s0 = kRegStart[r];
  const int ne = kRegStart[r + 1] - s0;
  const int j  = c - s0;
  int kj, row, k;
  if (o < 100)      { kj = 0; row = o;       k = 1; }
  else if (o < 150) { kj = 1; row = o - 100; k = 3; }
  else              { kj = 2; row = o - 150; k = 5; }
  const float* __restrict__ w = bp.w[r][kj];
  float s = bp.b[r][kj][row];
  const int pad = (k - 1) / 2;
  for (int dk = 0; dk < k; ++dk) {
    int jj = j + dk - pad;
    jj = ((jj % ne) + ne) % ne;               // circular ('wrap') padding
    const float* xr = x + (((size_t)b_ * 19 + (s0 + jj)) * 64 + t) * 200;
    const float* wr = w + (size_t)row * 200 * k + dk;
    float acc = 0.f;
    for (int i = 0; i < 200; ++i) acc += xr[i] * wr[(size_t)i * k];
    s += acc;
  }
  y[idx] = s + x[idx];
}

// ==================== attention reshapes ====================
// window: xw[((b*5+s)*19+c), n, f] = hp[b,c,n*5+s,f] (t=64 row zero-padded)
__global__ void window_gather(const float* __restrict__ ln, float* __restrict__ xw) {
  long idx = (long)blockIdx.x * blockDim.x + threadIdx.x;
  if (idx >= (long)TOKWn * 200) return;
  int f   = (int)(idx % 200);
  int nwi = (int)((idx / 200) % 13);
  int seq = (int)(idx / 2600);
  int c   = seq % 19;
  int s   = (seq / 19) % 5;
  int b_  = seq / 95;
  int t   = nwi * 5 + s;
  xw[idx] = (t < 64) ? ln[(((size_t)b_ * 19 + c) * 64 + t) * 200 + f] : 0.0f;
}

__global__ void window_scatter_add(const float* __restrict__ pr, float* __restrict__ cur) {
  long idx = (long)blockIdx.x * blockDim.x + threadIdx.x;
  if (idx >= TOKD) return;
  int f  = (int)(idx % 200);
  int t  = (int)((idx / 200) % 64);
  int c  = (int)((idx / 12800) % 19);
  int b_ = (int)(idx / 243200);
  int nwi = t / 5, s = t % 5;
  int seq = (b_ * 5 + s) * 19 + c;
  cur[idx] += pr[((size_t)seq * 13 + nwi) * 200 + f];
}

// region: broadcast per-region electrode mean of ln -> g (same [b,c,t,f] layout)
__global__ void region_mean_bcast(const float* __restrict__ ln, float* __restrict__ g) {
  long idx = (long)blockIdx.x * blockDim.x + threadIdx.x;
  if (idx >= TOKD) return;
  int f  = (int)(idx % 200);
  int t  = (int)((idx / 200) % 64);
  int c  = (int)((idx / 12800) % 19);
  int b_ = (int)(idx / 243200);
  int r = 0;
#pragma unroll
  for (int q = 0; q < 5; ++q) if (c >= kRegStart[q]) r = q;
  int s0 = kRegStart[r], e0 = kRegStart[r + 1];
  float s = 0.f;
  for (int cc = s0; cc < e0; ++cc) s += ln[(((size_t)b_ * 19 + cc) * 64 + t) * 200 + f];
  g[idx] = s / (float)(e0 - s0);
}

// xf[((b*64+t)*19+c), f] = xe[b,c,t,f]
__global__ void region_gather(const float* __restrict__ xe, float* __restrict__ xf) {
  long idx = (long)blockIdx.x * blockDim.x + threadIdx.x;
  if (idx >= TOKD) return;
  int f  = (int)(idx % 200);
  int c  = (int)((idx / 200) % 19);
  int t  = (int)((idx / 3800) % 64);
  int b_ = (int)(idx / 243200);
  xf[idx] = xe[(((size_t)b_ * 19 + c) * 64 + t) * 200 + f];
}

__global__ void region_scatter_add(const float* __restrict__ pr, float* __restrict__ cur) {
  long idx = (long)blockIdx.x * blockDim.x + threadIdx.x;
  if (idx >= TOKD) return;
  int f  = (int)(idx % 200);
  int t  = (int)((idx / 200) % 64);
  int c  = (int)((idx / 12800) % 19);
  int b_ = (int)(idx / 243200);
  cur[idx] += pr[(((size_t)b_ * 64 + t) * 19 + c) * 200 + f];
}

// ==================== host orchestration ====================
extern "C" void kernel_launch(void* const* d_in, const int* in_sizes, int n_in,
                              void* d_out, int out_size, void* d_ws, size_t ws_size,
                              hipStream_t stream) {
  (void)in_sizes; (void)n_in; (void)out_size; (void)ws_size;
  auto F = [&](int i) { return (const float*)d_in[i]; };
  const float* x   = F(0);
  const float* c1w = F(1),  *c1b = F(2),  *g1s = F(3),  *g1b = F(4);
  const float* c2w = F(5),  *c2b = F(6),  *g2s = F(7),  *g2b = F(8);
  const float* c3w = F(9),  *c3b = F(10), *g3s = F(11), *g3b = F(12);
  const float* specw = F(13), *specb = F(14), *posw = F(15), *posb = F(16);
  const float* t1w = F(17), *t1b = F(18), *t3w = F(19), *t3b = F(20), *t5w = F(21), *t5b = F(22);
  BrainParams bp;
  for (int r = 0; r < 5; ++r)
    for (int j = 0; j < 3; ++j) { bp.w[r][j] = F(23 + r * 6 + j * 2); bp.b[r][j] = F(23 + r * 6 + j * 2 + 1); }
  const float* ow = F(133), *ob = F(134);

  size_t off = 0;
  auto alloc = [&](size_t nf) {
    float* p = (float*)((char*)d_ws + off);
    off += ((nf * sizeof(float) + 255) / 256) * 256;
    return p;
  };
  float* bufA   = alloc(TOKD);                 // activations ping
  float* bufB   = alloc(TOKD);                 // activations pong
  float* bufLN  = alloc(TOKD);                 // layernorm output
  float* bufG   = alloc((size_t)TOKWn * 200);  // gathered seqs / gf broadcast
  float* bufQKV = alloc((size_t)TOKWn * 600);  // QKV (also conv scratch)
  float* bufAT  = alloc((size_t)TOKWn * 200);  // attention output
  float* bufPR  = alloc((size_t)TOKWn * 200);  // projection output
  float* bufH1  = alloc((size_t)TOKn * DFFn);  // FFN hidden (also conv scratch)
  float* bufSP  = alloc((size_t)TOKn * SPLD);  // spectrum (padded rows)
  float* bufTW  = alloc(101 * 200 * 2);        // DFT twiddles
  float* bufST  = alloc(256);                  // groupnorm stats

  dim3 b256(256);
  auto G1 = [](long n) { return dim3((unsigned)((n + 255) / 256)); };

  // ---------------- patch embedding ----------------
  patch_conv1<<<G1(3891200L), b256, 0, stream>>>(x, c1w, c1b, bufH1);
  gn_stats<<<dim3(80), b256, 0, stream>>>(bufH1, bufST);
  gn_apply_gelu<<<G1(3891200L), b256, 0, stream>>>(bufH1, bufST, g1s, g1b);
  patch_conv3x<<<G1(3891200L), b256, 0, stream>>>(bufH1, c2w, c2b, bufQKV);
  gn_stats<<<dim3(80), b256, 0, stream>>>(bufQKV, bufST);
  gn_apply_gelu<<<G1(3891200L), b256, 0, stream>>>(bufQKV, bufST, g2s, g2b);
  patch_conv3x<<<G1(3891200L), b256, 0, stream>>>(bufQKV, c3w, c3b, bufH1);
  gn_stats<<<dim3(80), b256, 0, stream>>>(bufH1, bufST);
  gn_apply_gelu<<<G1(3891200L), b256, 0, stream>>>(bufH1, bufST, g3s, g3b);
  patch_permute<<<G1(TOKD), b256, 0, stream>>>(bufH1, bufA);           // pe0 -> A
  twiddle_init<<<G1(101 * 200L), b256, 0, stream>>>(bufTW);
  dft_mag<<<G1((long)TOKn * 101), b256, 0, stream>>>(x, bufTW, bufSP);
  launch_gemm(false, bufSP, SPLD, specw, specb, bufA, bufB, TOKn, 200, 101, 0, stream); // pe1 -> B
  pos_conv_add<<<G1(TOKD), b256, 0, stream>>>(bufB, posw, posb, bufA); // pe -> A

  float* cur = bufA;
  float* tmp = bufB;

  // ---------------- 4 encoder layers ----------------
  for (int li = 0; li < 4; ++li) {
    const int base = 53 + li * 20;
    const float* n1s = F(base + 0),  *n1b = F(base + 1);
    const float* n2s = F(base + 2),  *n2b = F(base + 3);
    const float* n3s = F(base + 4),  *n3b = F(base + 5);
    const float* wwqkv = F(base + 6),  *wwb  = F(base + 7);
    const float* wwo   = F(base + 8),  *wwob = F(base + 9);
    const float* wrqkv = F(base + 10), *wrb  = F(base + 11);
    const float* wro   = F(base + 12), *wrob = F(base + 13);
    const float* gfw   = F(base + 14), *gfb  = F(base + 15);
    const float* l1w   = F(base + 16), *l1b  = F(base + 17);
    const float* l2w   = F(base + 18), *l2b  = F(base + 19);

    tem_embed_add<<<G1(TOKD), b256, 0, stream>>>(cur, tmp, t1w, t1b, t3w, t3b, t5w, t5b);
    { float* t_ = cur; cur = tmp; tmp = t_; }
    brain_embed_add<<<G1(TOKD), b256, 0, stream>>>(cur, tmp, bp);
    { float* t_ = cur; cur = tmp; tmp = t_; }

    // --- window attention (L=13, 1520 seqs) ---
    layernorm_rows<<<dim3(TOKn / 8), b256, 0, stream>>>(cur, n1s, n1b, bufLN, TOKn, 200);
    window_gather<<<G1((long)TOKWn * 200), b256, 0, stream>>>(bufLN, bufG);
    launch_gemm(true, bufG, 200, wwqkv, wwb, nullptr, bufQKV, TOKWn, 600, 200, 0, stream);
    mha_small<13><<<G1((long)SWINn * 13 * 8), b256, 0, stream>>>(bufQKV, bufAT, SWINn);
    launch_gemm(true, bufAT, 200, wwo, wwob, nullptr, bufPR, TOKWn, 200, 200, 0, stream);
    window_scatter_add<<<G1(TOKD), b256, 0, stream>>>(bufPR, cur);

    // --- region attention (L=19, 1024 seqs) ---
    layernorm_rows<<<dim3(TOKn / 8), b256, 0, stream>>>(cur, n2s, n2b, bufLN, TOKn, 200);
    region_mean_bcast<<<G1(TOKD), b256, 0, stream>>>(bufLN, bufG);
    launch_gemm(true, bufG, 200, gfw, gfb, bufLN, tmp, TOKn, 200, 200, 0, stream);  // xe -> tmp
    region_gather<<<G1(TOKD), b256, 0, stream>>>(tmp, bufG);
    launch_gemm(true, bufG, 200, wrqkv, wrb, nullptr, bufQKV, TOKn, 600, 200, 0, stream);
    mha_small<19><<<G1((long)SREGn * 19 * 8), b256, 0, stream>>>(bufQKV, bufAT, SREGn);
    launch_gemm(true, bufAT, 200, wro, wrob, nullptr, bufPR, TOKn, 200, 200, 0, stream);
    region_scatter_add<<<G1(TOKD), b256, 0, stream>>>(bufPR, cur);

    // --- FFN ---
    layernorm_rows<<<dim3(TOKn / 8), b256, 0, stream>>>(cur, n3s, n3b, bufLN, TOKn, 200);
    launch_gemm(false, bufLN, 200, l1w, l1b, nullptr, bufH1, TOKn, DFFn, 200, 1, stream);
    launch_gemm(false, bufH1, DFFn, l2w, l2b, cur, tmp, TOKn, 200, DFFn, 0, stream);
    { float* t_ = cur; cur = tmp; tmp = t_; }
  }

  // ---------------- output projection ----------------
  launch_gemm(false, cur, 200, ow, ob, nullptr, (float*)d_out, TOKn, 200, 200, 0, stream);
}
